// CrossAttention_42786464202926
// MI455X (gfx1250) — matrix operand order repaired and checked by reading, not compiled, not run on hardware
//
#include <hip/hip_runtime.h>
#include <math.h>

// Problem constants (from reference)
constexpr int kB   = 8;
constexpr int kSQ  = 2048;
constexpr int kSK  = 2048;
constexpr int kDim = 1024;

typedef __attribute__((ext_vector_type(16))) __bf16          v16bf;
typedef __attribute__((ext_vector_type(16))) unsigned short  v16u;
typedef __attribute__((ext_vector_type(8)))  unsigned short  v8u;
typedef __attribute__((ext_vector_type(8)))  float           v8f;
typedef __attribute__((ext_vector_type(4)))  float           v4f;

static __device__ __forceinline__ unsigned short f2bf(float f) {
  unsigned int u = __builtin_bit_cast(unsigned int, f);
  u += 0x7FFFu + ((u >> 16) & 1u);            // round-to-nearest-even
  return (unsigned short)(u >> 16);
}

static __device__ __forceinline__ v8f v8f_zero() {
  v8f z;
#pragma unroll
  for (int i = 0; i < 8; ++i) z[i] = 0.f;
  return z;
}

static __device__ __forceinline__ v8f wmma_bf16(v16u a, v16u b, v8f c) {
  return __builtin_amdgcn_wmma_f32_16x16x32_bf16(
      false, __builtin_bit_cast(v16bf, a),
      false, __builtin_bit_cast(v16bf, b),
      (short)0, c, false, false);
}

// ---------------------------------------------------------------------------
// Projection GEMM: Y = X[M,K] * W[N,K]^T + bias, output bf16.
// If TRANS_OUT, store per-batch transposed: Y[b][n][t] (for V).
// M = kB*kSK = 16384, N = K = kDim = 1024. Tile 128x128x32, 8 waves.
// ---------------------------------------------------------------------------
template <bool TRANS_OUT>
__global__ __launch_bounds__(256)
void proj_gemm_kernel(const float* __restrict__ X, const float* __restrict__ W,
                      const float* __restrict__ bias,
                      unsigned short* __restrict__ Y) {
  constexpr int N = kDim, K = kDim;

  __shared__ unsigned short As[128][40];  // [m][k] bf16, padded
  __shared__ unsigned short Bs[128][40];  // [n][k] bf16 (= W rows, no transpose)

  const int tid = threadIdx.x;
  const int lane = tid & 31, lr = lane & 15, hl = lane >> 4;
  const int w  = tid >> 5;
  const int wm = w >> 1;               // 0..3 : 32-row slice
  const int wn = w & 1;                // 0..1 : 64-col slice
  const int bm = blockIdx.x;           // 0..127
  const int bn = blockIdx.y;           // 0..7

  v8f acc[2][4];
#pragma unroll
  for (int i = 0; i < 2; ++i)
#pragma unroll
    for (int j = 0; j < 4; ++j) acc[i][j] = v8f_zero();

  const int tcol = (tid & 7) * 4;      // 0..28, float4 per thread
  const int trow = tid >> 3;           // 0..31

  for (int k0 = 0; k0 < K; k0 += 32) {
    // ---- stage A (x rows) and B (W rows) as bf16 into LDS ----
#pragma unroll
    for (int rr = 0; rr < 4; ++rr) {
      const int r = trow + rr * 32;
      v4f xa = *(const v4f*)(X + (size_t)(bm * 128 + r) * K + k0 + tcol);
      v4f wa = *(const v4f*)(W + (size_t)(bn * 128 + r) * K + k0 + tcol);
      union { unsigned short s[4]; unsigned int u[2]; } pa, pb;
#pragma unroll
      for (int j = 0; j < 4; ++j) { pa.s[j] = f2bf(xa[j]); pb.s[j] = f2bf(wa[j]); }
      *(unsigned int*)&As[r][tcol]     = pa.u[0];
      *(unsigned int*)&As[r][tcol + 2] = pa.u[1];
      *(unsigned int*)&Bs[r][tcol]     = pb.u[0];
      *(unsigned int*)&Bs[r][tcol + 2] = pb.u[1];
    }
    __syncthreads();

    // ---- fragments in contiguous k order (k = 16*hl + j) for A and B, then WMMA ----
    v16u a[2];
#pragma unroll
    for (int mt = 0; mt < 2; ++mt) {
      const unsigned short* ap = &As[wm * 32 + mt * 16 + lr][hl * 16];
#pragma unroll
      for (int j = 0; j < 8; ++j) { a[mt][j] = ap[j]; a[mt][8 + j] = ap[8 + j]; }
    }
#pragma unroll
    for (int nt = 0; nt < 4; ++nt) {
      const unsigned short* bp = &Bs[wn * 64 + nt * 16 + lr][hl * 16];
      v16u bf;
#pragma unroll
      for (int j = 0; j < 16; ++j) bf[j] = bp[j];
      acc[0][nt] = wmma_bf16(a[0], bf, acc[0][nt]);
      acc[1][nt] = wmma_bf16(a[1], bf, acc[1][nt]);
    }
    __syncthreads();
  }

  // ---- epilogue: bias add, bf16 store ----
#pragma unroll
  for (int mt = 0; mt < 2; ++mt) {
    const int mbase = bm * 128 + wm * 32 + mt * 16 + hl * 8;  // 8 consecutive rows
#pragma unroll
    for (int nt = 0; nt < 4; ++nt) {
      const int n = bn * 128 + wn * 64 + nt * 16 + lr;
      const float bv = bias[n];
      if (TRANS_OUT) {
        v8u pk;
#pragma unroll
        for (int r = 0; r < 8; ++r) pk[r] = f2bf(acc[mt][nt][r] + bv);
        const int batch = mbase / kSK;
        const int t0    = mbase % kSK;
        *(v8u*)&Y[((size_t)batch * kDim + n) * kSK + t0] = pk;  // Y[b][n][t]
      } else {
#pragma unroll
        for (int r = 0; r < 8; ++r)
          Y[(size_t)(mbase + r) * kDim + n] = f2bf(acc[mt][nt][r] + bv);
      }
    }
  }
}

// ---------------------------------------------------------------------------
// Flash attention: one WG per (batch, 16-query tile). 8 waves.
// Q bf16 [b][s][d], K bf16 [b][t][d], V bf16 TRANSPOSED [b][d][t].
// ---------------------------------------------------------------------------
__global__ __launch_bounds__(256)
void attn_kernel(const unsigned short* __restrict__ Qb,
                 const unsigned short* __restrict__ Kb,
                 const unsigned short* __restrict__ Vt,
                 float* __restrict__ Out) {
  __shared__ unsigned short Qs[16][1032];   // 33 KB: query tile, bf16
  __shared__ float          Ss[16][132];    // 8.4 KB: raw scores
  __shared__ unsigned short Ps[16][128];    // 4 KB: softmax probs, bf16
  __shared__ float          red[16][16];
  __shared__ float          mrow[16], lrow[16], arow[16];

  const int tid = threadIdx.x;
  const int lane = tid & 31, lr = lane & 15, hl = lane >> 4;
  const int w  = tid >> 5;
  const int b  = blockIdx.y;
  const int q0 = blockIdx.x * 16;

  const unsigned short* Q = Qb + ((size_t)b * kSQ + q0) * kDim;
  const unsigned short* K = Kb + (size_t)b * kSK * kDim;
  const unsigned short* V = Vt + (size_t)b * kDim * kSK;
  float* O = Out + ((size_t)b * kSQ + q0) * kDim;

  {  // load Q tile into LDS (16B vectors)
    const int qr = tid >> 4, qcb = (tid & 15) * 8;
#pragma unroll
    for (int p = 0; p < 8; ++p) {
      const int c = qcb + p * 128;
      *(v8u*)&Qs[qr][c] = *(const v8u*)&Q[(size_t)qr * kDim + c];
    }
  }
  if (tid < 16) { mrow[tid] = -INFINITY; lrow[tid] = 0.f; }

  v8f oacc[8];
#pragma unroll
  for (int nt = 0; nt < 8; ++nt) oacc[nt] = v8f_zero();

  __syncthreads();

  const int srow = tid >> 4, st = tid & 15;

  for (int kt = 0; kt < kSK / 128; ++kt) {
    const int key0 = kt * 128;

    // ---- S tile: this wave computes 16 key columns, full D reduction ----
    const int keyrow = key0 + w * 16 + lr;
    const unsigned short* kp = &K[(size_t)keyrow * kDim + hl * 16];
    if (kt + 1 < kSK / 128)
      __builtin_prefetch(kp + (size_t)128 * kDim, 0, 1);  // next key tile
    v8f s = v8f_zero();
    for (int d0 = 0; d0 < kDim; d0 += 32) {
      v16u a, bf;
      const unsigned short* ap = &Qs[lr][d0 + hl * 16];
#pragma unroll
      for (int j = 0; j < 8; ++j) { a[j] = ap[j]; a[8 + j] = ap[8 + j]; }
      const unsigned short* bp = kp + d0;
#pragma unroll
      for (int j = 0; j < 16; ++j) bf[j] = bp[j];
      s = wmma_bf16(a, bf, s);
    }
#pragma unroll
    for (int r = 0; r < 8; ++r) Ss[r + hl * 8][w * 16 + lr] = s[r] * 0.03125f;
    __syncthreads();

    // ---- online softmax (16 threads per row) ----
    float lm = -INFINITY;
#pragma unroll
    for (int c = 0; c < 8; ++c) lm = fmaxf(lm, Ss[srow][st * 8 + c]);
    red[srow][st] = lm;
    __syncthreads();
    if (st == 0) {
      float mt = red[srow][0];
#pragma unroll
      for (int j = 1; j < 16; ++j) mt = fmaxf(mt, red[srow][j]);
      const float mo = mrow[srow];
      const float mn = fmaxf(mo, mt);
      arow[srow] = __expf(mo - mn);
      mrow[srow] = mn;
    }
    __syncthreads();
    const float mn = mrow[srow];
    float ls = 0.f;
#pragma unroll
    for (int c = 0; c < 8; ++c) {
      const float p = __expf(Ss[srow][st * 8 + c] - mn);
      Ps[srow][st * 8 + c] = f2bf(p);
      ls += p;
    }
    red[srow][st] = ls;
    __syncthreads();
    if (st == 0) {
      float sum = 0.f;
#pragma unroll
      for (int j = 0; j < 16; ++j) sum += red[srow][j];
      lrow[srow] = lrow[srow] * arow[srow] + sum;
    }

    // ---- O = O*alpha + P @ V ; this wave owns 128 output columns ----
    float alr[8];
#pragma unroll
    for (int r = 0; r < 8; ++r) alr[r] = arow[r + hl * 8];
#pragma unroll
    for (int nt = 0; nt < 8; ++nt)
#pragma unroll
      for (int r = 0; r < 8; ++r) oacc[nt][r] *= alr[r];

#pragma unroll
    for (int kk = 0; kk < 128; kk += 32) {
      v16u a;
      const unsigned short* ap = &Ps[lr][kk + hl * 16];
#pragma unroll
      for (int j = 0; j < 8; ++j) { a[j] = ap[j]; a[8 + j] = ap[8 + j]; }
#pragma unroll
      for (int nt = 0; nt < 8; ++nt) {
        const int e = w * 128 + nt * 16 + lr;
        const unsigned short* bp = &V[(size_t)e * kSK + key0 + kk + hl * 16];
        v16u bf;
#pragma unroll
        for (int j = 0; j < 16; ++j) bf[j] = bp[j];
        oacc[nt] = wmma_bf16(a, bf, oacc[nt]);
      }
    }
    __syncthreads();
  }

  // ---- normalize and store fp32 output ----
  float invl[8];
#pragma unroll
  for (int r = 0; r < 8; ++r) invl[r] = 1.0f / lrow[r + hl * 8];
#pragma unroll
  for (int nt = 0; nt < 8; ++nt) {
    const int col = w * 128 + nt * 16 + lr;
#pragma unroll
    for (int r = 0; r < 8; ++r)
      O[(size_t)(r + hl * 8) * kDim + col] = oacc[nt][r] * invl[r];
  }
}

// ---------------------------------------------------------------------------
extern "C" void kernel_launch(void* const* d_in, const int* in_sizes, int n_in,
                              void* d_out, int out_size, void* d_ws, size_t ws_size,
                              hipStream_t stream) {
  const float* x   = (const float*)d_in[0];
  const float* ctx = (const float*)d_in[1];
  const float* Wq  = (const float*)d_in[2];
  const float* bq  = (const float*)d_in[3];
  const float* Wk  = (const float*)d_in[4];
  const float* bk  = (const float*)d_in[5];
  const float* Wv  = (const float*)d_in[6];
  const float* bv  = (const float*)d_in[7];
  float* out = (float*)d_out;

  const size_t elems = (size_t)kB * kSK * kDim;  // 16M elements each
  unsigned short* Qb = (unsigned short*)d_ws;
  unsigned short* Kb = Qb + elems;
  unsigned short* Vb = Kb + elems;  // stored transposed [b][d][t]

  dim3 gproj(kB * kSQ / 128, kDim / 128);  // (128, 8)
  proj_gemm_kernel<false><<<gproj, 256, 0, stream>>>(x,   Wq, bq, Qb);
  proj_gemm_kernel<false><<<gproj, 256, 0, stream>>>(ctx, Wk, bk, Kb);
  proj_gemm_kernel<true ><<<gproj, 256, 0, stream>>>(ctx, Wv, bv, Vb);

  dim3 gattn(kSQ / 16, kB);                // (128, 8)
  attn_kernel<<<gattn, 256, 0, stream>>>(Qb, Kb, Vb, out);
}
